// ImageEncoderViT_13864154432268
// MI455X (gfx1250) — compile-verified, matrix operand order checked by reading, not run on hardware
//
#include <hip/hip_runtime.h>
#include <math.h>

// ---------------------------------------------------------------------------
// MI455X (gfx1250) ViT encoder. wave32, WMMA f32_16x16x32_f16.
// fp32 weights converted to f16 in registers inside the GEMM (single HBM pass
// over ~115MB of weights == the 23.3TB/s roofline limiter). Hot loops are
// branch-free; scattered-B weights are read through 16 independent address
// registers with immediate per-subtile offsets so loads clause without waits.
// ---------------------------------------------------------------------------

typedef __attribute__((ext_vector_type(16))) _Float16 v16h;
typedef __attribute__((ext_vector_type(8)))  float    v8f;

#define TOKENS 864      // 6*12*12
#define EMB    768
#define NHEADS 12
#define HD     64

__device__ __forceinline__ float gelu_exact(float x) {
  return 0.5f * x * (1.0f + erff(x * 0.70710678118654752f));
}

// ISA 7.12.2 wave32 16-bit fragment K index for v16h element e, half-group kg:
// e 0..7 -> k = kg*8 + e ; e 8..15 -> k = 16 + kg*8 + (e-8)
__device__ __forceinline__ int kpat(int e, int kg) {
  return kg * 8 + (e & 7) + ((e >> 3) << 4);
}

__device__ __forceinline__ v16h cvt16(float4 a0, float4 a1, float4 a2, float4 a3) {
  v16h f;
  f[0]=(_Float16)a0.x;  f[1]=(_Float16)a0.y;  f[2]=(_Float16)a0.z;  f[3]=(_Float16)a0.w;
  f[4]=(_Float16)a1.x;  f[5]=(_Float16)a1.y;  f[6]=(_Float16)a1.z;  f[7]=(_Float16)a1.w;
  f[8]=(_Float16)a2.x;  f[9]=(_Float16)a2.y;  f[10]=(_Float16)a2.z; f[11]=(_Float16)a2.w;
  f[12]=(_Float16)a3.x; f[13]=(_Float16)a3.y; f[14]=(_Float16)a3.z; f[15]=(_Float16)a3.w;
  return f;
}

// ---------------------------------------------------------------------------
// Batched WMMA GEMM:  C[z] = act( alpha * A[z] x B[z] + bias )
//   A: (M,K) at A + z*sA + m*lda + k            (k contiguous, always)
//   B: (K,N) at B + z*sB + k*ldbk + n*ldbn      (strides allow B^T)
//   C: (M,N) at C + z*sC + m*ldc + n
// One wave computes a 16x64 strip (4 accumulators, A fragment reused 4x).
// Contract used by the scattered-B path (ldbk != 1): ldbn == 1 and N % 64 == 0
// (true for all call sites below: qkv/proj/fc1/fc2 weights and attn.V).
// ---------------------------------------------------------------------------
__global__ void __launch_bounds__(128)
k_gemm_wmma(const float* __restrict__ A, long lda, long sA,
            const float* __restrict__ B, long ldbk, long ldbn, long sB,
            float* C, long ldc, long sC,
            int M, int N, int K, float alpha,
            const float* __restrict__ bias, int act)
{
  const int lane = threadIdx.x & 31;
  const int wave = threadIdx.x >> 5;
  const int tn0  = (blockIdx.x * 4 + wave) * 4;   // first 16-wide tile of strip
  if (tn0 * 16 >= N) return;                      // idle wave (no barriers)
  const int tm = blockIdx.y;
  const int z  = blockIdx.z;

  const float* Ab = A + (size_t)z * sA;
  const float* Bb = B + (size_t)z * sB;
  float*       Cb = C + (size_t)z * sC;

  const int ml = lane & 15;
  const int kg = lane >> 4;
  // Clamped row: OOB A rows only affect outputs we never store.
  const int mrow = min(tm * 16 + ml, M - 1);

  v8f acc[4];
#pragma unroll
  for (int t = 0; t < 4; ++t) acc[t] = (v8f){0.f,0.f,0.f,0.f,0.f,0.f,0.f,0.f};

  const int K32 = K & ~31;
  const float* ap = Ab + (size_t)mrow * lda + kg * 8;   // run0; run1 at +16

  if (ldbk == 1) {
    // ---- B is k-contiguous: both operands load as float4 runs -------------
    int ncol[4];
#pragma unroll
    for (int t = 0; t < 4; ++t) ncol[t] = min((tn0 + t) * 16 + ml, N - 1);
    const float* bp[4];
#pragma unroll
    for (int t = 0; t < 4; ++t) bp[t] = Bb + (size_t)ncol[t] * ldbn + kg * 8;
    for (int kb = 0; kb < K32; kb += 32) {
      v16h af = cvt16(*(const float4*)(ap + kb),      *(const float4*)(ap + kb + 4),
                      *(const float4*)(ap + kb + 16), *(const float4*)(ap + kb + 20));
#pragma unroll
      for (int t = 0; t < 4; ++t) {
        if ((tn0 + t) * 16 >= N) break;            // uniform scalar branch
        v16h bf = cvt16(*(const float4*)(bp[t] + kb),      *(const float4*)(bp[t] + kb + 4),
                        *(const float4*)(bp[t] + kb + 16), *(const float4*)(bp[t] + kb + 20));
        acc[t] = __builtin_amdgcn_wmma_f32_16x16x32_f16(false, af, false, bf,
                                                        (short)0, acc[t], false, false);
      }
    }
  } else {
    // ---- B strided in k, n-contiguous (ldbn==1, N%64==0 guaranteed) ------
    // 16 independent base addresses; per-subtile n offset (t*16 floats) is a
    // compile-time immediate folded into the load; one stride add per iter.
    const float* bpe[16];
#pragma unroll
    for (int e = 0; e < 16; ++e)
      bpe[e] = Bb + (size_t)kpat(e, kg) * ldbk + (tn0 * 16 + ml);
    const size_t bstep = (size_t)32 * ldbk;
    for (int kb = 0; kb < K32; kb += 32) {
      v16h af = cvt16(*(const float4*)(ap + kb),      *(const float4*)(ap + kb + 4),
                      *(const float4*)(ap + kb + 16), *(const float4*)(ap + kb + 20));
#pragma unroll
      for (int t = 0; t < 4; ++t) {
        v16h bf;
#pragma unroll
        for (int e = 0; e < 16; ++e)
          bf[e] = (_Float16)bpe[e][t * 16];        // imm offset 0/64/128/192B
        acc[t] = __builtin_amdgcn_wmma_f32_16x16x32_f16(false, af, false, bf,
                                                        (short)0, acc[t], false, false);
      }
#pragma unroll
      for (int e = 0; e < 16; ++e) bpe[e] += bstep;
    }
  }

  // ---- K tail (only K%32 != 0, e.g. attn.V with K=216): branch-free -------
  if (K32 < K) {
    v16h af;
#pragma unroll
    for (int e = 0; e < 16; ++e) {
      int k  = K32 + kpat(e, kg);
      int kc = min(k, K - 1);
      float av = Ab[(size_t)mrow * lda + kc];
      af[e] = (_Float16)((k < K) ? av : 0.f);
    }
#pragma unroll
    for (int t = 0; t < 4; ++t) {
      if ((tn0 + t) * 16 >= N) break;
      const int nc = min((tn0 + t) * 16 + ml, N - 1);
      v16h bf;
#pragma unroll
      for (int e = 0; e < 16; ++e) {
        int k  = K32 + kpat(e, kg);
        int kc = min(k, K - 1);
        float bv = Bb[(size_t)kc * ldbk + (size_t)nc * ldbn];
        bf[e] = (_Float16)((k < K) ? bv : 0.f);
      }
      acc[t] = __builtin_amdgcn_wmma_f32_16x16x32_f16(false, af, false, bf,
                                                      (short)0, acc[t], false, false);
    }
  }

  // ---- epilogue: C layout m = r + 8*kg, n = ml ----------------------------
#pragma unroll
  for (int t = 0; t < 4; ++t) {
    if ((tn0 + t) * 16 >= N) break;
    const int n = (tn0 + t) * 16 + ml;
#pragma unroll
    for (int r = 0; r < 8; ++r) {
      const int m = tm * 16 + r + kg * 8;
      if (m < M && n < N) {
        float v = alpha * acc[t][r];
        if (bias) v += bias[n];
        if (act == 1) v = gelu_exact(v);
        Cb[(size_t)m * ldc + n] = v;
      }
    }
  }
}

// --------------------------- elementwise helpers ---------------------------
__global__ void k_add_inplace(float* x, const float* __restrict__ y, int n) {
  int i = blockIdx.x * blockDim.x + threadIdx.x;
  if (i < n) x[i] += y[i];
}

// patch im2col: (864, 4096) rows match patch_w (768,1,16,16,16) K order
__global__ void k_im2col_patch(const float* __restrict__ x, float* __restrict__ col) {
  int i = blockIdx.x * blockDim.x + threadIdx.x;
  if (i >= TOKENS * 4096) return;
  int t = i >> 12, j = i & 4095;
  int d = t / 144, h = (t / 12) % 12, w = t % 12;
  int kd = j >> 8, kh = (j >> 4) & 15, kw = j & 15;
  col[i] = x[(size_t)(d * 16 + kd) * 192 * 192 + (h * 16 + kh) * 192 + (w * 16 + kw)];
}

// window partition gather: dst (window order) <- src (grid order). ws=6, D=6,H=W=12.
__global__ void k_win_gather(const float* __restrict__ src, float* __restrict__ dst) {
  int i = blockIdx.x * blockDim.x + threadIdx.x;
  if (i >= TOKENS * EMB) return;
  int tw = i / EMB, c = i % EMB;
  int win = tw / 216, l = tw % 216;
  int hb = win / 2, wb = win % 2;
  int dd = l / 36, hh = (l / 6) % 6, ww = l % 6;
  int g = dd * 144 + (hb * 6 + hh) * 12 + (wb * 6 + ww);
  dst[i] = src[(size_t)g * EMB + c];
}

// window unpartition + residual:  x(grid) += proj(window order)
__global__ void k_win_scatter_add(float* x, const float* __restrict__ proj) {
  int i = blockIdx.x * blockDim.x + threadIdx.x;
  if (i >= TOKENS * EMB) return;
  int g = i / EMB, c = i % EMB;
  int d = g / 144, h = (g / 12) % 12, w = g % 12;
  int tw = (h / 6 * 2 + w / 6) * 216 + d * 36 + (h % 6) * 6 + (w % 6);
  x[i] += proj[(size_t)tw * EMB + c];
}

// Row LayerNorm (also channel-LN in token-major layout). blockDim=256.
__global__ void k_layernorm(const float* __restrict__ src, float* __restrict__ dst,
                            const float* __restrict__ w, const float* __restrict__ b,
                            int C, float eps) {
  const int row = blockIdx.x;
  const float* s = src + (size_t)row * C;
  float* d = dst + (size_t)row * C;
  __shared__ float red[256];
  float acc = 0.f;
  for (int c = threadIdx.x; c < C; c += 256) acc += s[c];
  red[threadIdx.x] = acc; __syncthreads();
  for (int st = 128; st > 0; st >>= 1) {
    if (threadIdx.x < st) red[threadIdx.x] += red[threadIdx.x + st];
    __syncthreads();
  }
  const float mean = red[0] / C; __syncthreads();
  acc = 0.f;
  for (int c = threadIdx.x; c < C; c += 256) { float t = s[c] - mean; acc += t * t; }
  red[threadIdx.x] = acc; __syncthreads();
  for (int st = 128; st > 0; st >>= 1) {
    if (threadIdx.x < st) red[threadIdx.x] += red[threadIdx.x + st];
    __syncthreads();
  }
  const float inv = rsqrtf(red[0] / C + eps);
  for (int c = threadIdx.x; c < C; c += 256)
    d[c] = (s[c] - mean) * inv * w[c] + b[c];
}

// Decomposed rel-pos query dots: rel[(z*Nt+n)*32 + {k | S0+k | S0+S1+k}] = q.R
__global__ void k_relq(const float* __restrict__ qkv, int Nt,
                       int S0, int S1, int S2,
                       const float* __restrict__ tabD, const float* __restrict__ tabH,
                       const float* __restrict__ tabW, float* __restrict__ rel) {
  const int n = blockIdx.x, z = blockIdx.y;
  const int win = z / NHEADS, head = z % NHEADS;
  const float* q = qkv + (size_t)(win * Nt + n) * (3 * EMB) + head * HD;
  __shared__ float sq[HD];
  if (threadIdx.x < HD) sq[threadIdx.x] = q[threadIdx.x];
  __syncthreads();
  const int t = threadIdx.x, total = S0 + S1 + S2;
  if (t >= total) return;
  const int dq = n / (S1 * S2), hq = (n / S2) % S1, wq = n % S2;
  const float* tab; int k, S, coord;
  if (t < S0)            { tab = tabD; k = t;           S = S0; coord = dq; }
  else if (t < S0 + S1)  { tab = tabH; k = t - S0;      S = S1; coord = hq; }
  else                   { tab = tabW; k = t - S0 - S1; S = S2; coord = wq; }
  const float* r = tab + (size_t)(coord - k + S - 1) * HD;
  float s = 0.f;
#pragma unroll
  for (int c = 0; c < HD; ++c) s += sq[c] * r[c];
  rel[((size_t)z * Nt + n) * 32 + t] = s;
}

// softmax over keys with decomposed rel-pos bias added. one block per row.
__global__ void k_softmax_bias(float* attn, const float* __restrict__ rel,
                               int Nt, int S0, int S1, int S2) {
  const int row = blockIdx.x;
  const int z = row / Nt, n = row % Nt;
  float* a = attn + (size_t)row * Nt;
  const float* r = rel + ((size_t)z * Nt + n) * 32;
  __shared__ float red[256];
  float vals[4];
  float mx = -3.4e38f;
#pragma unroll
  for (int it = 0; it < 4; ++it) {
    int m = threadIdx.x + it * 256;
    if (m < Nt) {
      int dm = m / (S1 * S2), hm = (m / S2) % S1, wm = m % S2;
      float v = a[m] + r[dm] + r[S0 + hm] + r[S0 + S1 + wm];
      vals[it] = v; mx = fmaxf(mx, v);
    }
  }
  red[threadIdx.x] = mx; __syncthreads();
  for (int st = 128; st > 0; st >>= 1) {
    if (threadIdx.x < st) red[threadIdx.x] = fmaxf(red[threadIdx.x], red[threadIdx.x + st]);
    __syncthreads();
  }
  mx = red[0]; __syncthreads();
  float sum = 0.f;
#pragma unroll
  for (int it = 0; it < 4; ++it) {
    int m = threadIdx.x + it * 256;
    if (m < Nt) { vals[it] = expf(vals[it] - mx); sum += vals[it]; }
  }
  red[threadIdx.x] = sum; __syncthreads();
  for (int st = 128; st > 0; st >>= 1) {
    if (threadIdx.x < st) red[threadIdx.x] += red[threadIdx.x + st];
    __syncthreads();
  }
  const float inv = 1.0f / red[0];
#pragma unroll
  for (int it = 0; it < 4; ++it) {
    int m = threadIdx.x + it * 256;
    if (m < Nt) a[m] = vals[it] * inv;
  }
}

// 3x3x3 im2col with zero pad: col (864, 6912); K order = ci*27 + kd*9 + kh*3 + kw
__global__ void k_im2col_neck(const float* __restrict__ src, float* __restrict__ col) {
  int i = blockIdx.x * blockDim.x + threadIdx.x;
  if (i >= TOKENS * 6912) return;
  int t = i / 6912, j = i % 6912;
  int ci = j / 27, rr = j % 27;
  int kd = rr / 9, kh = (rr / 3) % 3, kw = rr % 3;
  int d = t / 144, h = (t / 12) % 12, w = t % 12;
  int dd = d + kd - 1, hh = h + kh - 1, ww = w + kw - 1;
  float v = 0.f;
  if (dd >= 0 && dd < 6 && hh >= 0 && hh < 12 && ww >= 0 && ww < 12)
    v = src[(size_t)(dd * 144 + hh * 12 + ww) * 256 + ci];
  col[i] = v;
}

// Final channel-LN writing NCDHW-transposed output. blockDim = 256 (= C).
__global__ void k_ln_ch_out(const float* __restrict__ src, const float* __restrict__ w,
                            const float* __restrict__ b, float* __restrict__ out) {
  const int t = blockIdx.x, c = threadIdx.x;
  __shared__ float red[256];
  const float v = src[(size_t)t * 256 + c];
  red[c] = v; __syncthreads();
  for (int st = 128; st > 0; st >>= 1) {
    if (c < st) red[c] += red[c + st];
    __syncthreads();
  }
  const float mean = red[0] / 256.f; __syncthreads();
  float dsq = (v - mean) * (v - mean);
  red[c] = dsq; __syncthreads();
  for (int st = 128; st > 0; st >>= 1) {
    if (c < st) red[c] += red[c + st];
    __syncthreads();
  }
  const float inv = rsqrtf(red[0] / 256.f + 1e-6f);
  out[(size_t)c * TOKENS + t] = (v - mean) * inv * w[c] + b[c];
}

// ---------------------------------------------------------------------------
static inline void gemm(hipStream_t s,
                        const float* A, long lda, long sA,
                        const float* B, long ldbk, long ldbn, long sB,
                        float* C, long ldc, long sC,
                        int M, int N, int K, int batch, float alpha,
                        const float* bias, int act) {
  int tilesM = (M + 15) / 16;
  int strips = (N + 63) / 64;                 // 16x64 strip per wave
  dim3 g((strips + 3) / 4, tilesM, batch), b(128);
  hipLaunchKernelGGL(k_gemm_wmma, g, b, 0, s, A, lda, sA, B, ldbk, ldbn, sB,
                     C, ldc, sC, M, N, K, alpha, bias, act);
}

extern "C" void kernel_launch(void* const* d_in, const int* in_sizes, int n_in,
                              void* d_out, int out_size, void* d_ws, size_t ws_size,
                              hipStream_t stream) {
  (void)in_sizes; (void)n_in; (void)out_size; (void)ws_size;
  auto in = [&](int i) { return (const float*)d_in[i]; };

  const float* X       = in(0);   // (96,192,192)
  const float* patch_w = in(1);   // (768, 4096)
  const float* patch_b = in(2);   // (768)
  const float* pos     = in(3);   // (864, 768)

  struct Blk {
    const float *n1w, *n1b, *qkvw, *qkvb, *pw, *pb, *n2w, *n2b,
                *f1w, *f1b, *f2w, *f2b, *rd, *rh, *rw;
  } blk[4];
  for (int i = 0; i < 4; ++i) {
    int o = 4 + i * 15;
    blk[i] = { in(o+0), in(o+1), in(o+2), in(o+3), in(o+4), in(o+5), in(o+6),
               in(o+7), in(o+8), in(o+9), in(o+10), in(o+11), in(o+12),
               in(o+13), in(o+14) };
  }
  const float* neck_w1  = in(64); // (256, 768)
  const float* neck_l1w = in(65);
  const float* neck_l1b = in(66);
  const float* neck_w2  = in(67); // (256, 6912)
  const float* neck_l2w = in(68);
  const float* neck_l2b = in(69);

  // workspace layout (floats). regionA is time-shared by im2col / attn / MLP.
  float* ws = (float*)d_ws;
  size_t off = 0;
  float* regionA = ws + off; off += 9000000;            // max: 12*864*864 attn
  float* x   = ws + off; off += TOKENS * EMB;           // residual stream
  float* t1  = ws + off; off += TOKENS * EMB;           // LN out
  float* t2  = ws + off; off += TOKENS * EMB;           // win-gather / attn-out
  float* t3  = ws + off; off += TOKENS * EMB;           // proj / fc2 out
  float* qkv = ws + off; off += (size_t)TOKENS * 3 * EMB;
  float* rel = ws + off; off += (size_t)NHEADS * TOKENS * 32;
  float* na  = ws + off; off += TOKENS * 256;
  float* nb  = ws + off; off += TOKENS * 256;

  const int TPB = 256;
  auto nblk = [](long n) { return (unsigned)((n + 255) / 256); };

  // ---- patch embed: im2col + GEMM(+bias), then + pos_embed ----------------
  hipLaunchKernelGGL(k_im2col_patch, dim3(nblk((long)TOKENS * 4096)), dim3(TPB), 0, stream,
                     X, regionA);
  gemm(stream, regionA, 4096, 0, patch_w, 1, 4096, 0, x, EMB, 0,
       TOKENS, EMB, 4096, 1, 1.f, patch_b, 0);
  hipLaunchKernelGGL(k_add_inplace, dim3(nblk(TOKENS * EMB)), dim3(TPB), 0, stream,
                     x, pos, TOKENS * EMB);

  // ---- transformer blocks -------------------------------------------------
  for (int i = 0; i < 4; ++i) {
    const Blk& p = blk[i];
    const bool glob = (i == 3);
    const int nw = glob ? 1 : 4;
    const int Nt = glob ? TOKENS : 216;
    const int S0 = 6, S1 = glob ? 12 : 6, S2 = glob ? 12 : 6;

    // LN1 (+ window partition)
    hipLaunchKernelGGL(k_layernorm, dim3(TOKENS), dim3(TPB), 0, stream,
                       x, t1, p.n1w, p.n1b, EMB, 1e-5f);
    const float* src = t1;
    if (!glob) {
      hipLaunchKernelGGL(k_win_gather, dim3(nblk(TOKENS * EMB)), dim3(TPB), 0, stream, t1, t2);
      src = t2;
    }

    // qkv projection: (864,768) x (768,2304) + bias
    gemm(stream, src, EMB, 0, p.qkvw, 3 * EMB, 1, 0, qkv, 3 * EMB, 0,
         TOKENS, 3 * EMB, EMB, 1, 1.f, p.qkvb, 0);

    // scores: per window, batched over heads. S = (q/sqrt(d)) . K^T
    float* attn = regionA;
    for (int w = 0; w < nw; ++w) {
      const float* qb = qkv + (size_t)w * Nt * 3 * EMB;
      gemm(stream, qb,          3 * EMB, HD,          // A = Q  (head stride 64)
           qb + EMB,  1, 3 * EMB, HD,                 // B = K^T (k-contiguous)
           attn + (size_t)w * NHEADS * Nt * Nt, Nt, (long)Nt * Nt,
           Nt, Nt, HD, NHEADS, 0.125f, nullptr, 0);
    }

    // rel-pos query dots + bias-fused softmax
    hipLaunchKernelGGL(k_relq, dim3(Nt, nw * NHEADS), dim3(64), 0, stream,
                       qkv, Nt, S0, S1, S2, p.rd, p.rh, p.rw, rel);
    hipLaunchKernelGGL(k_softmax_bias, dim3(nw * NHEADS * Nt), dim3(TPB), 0, stream,
                       attn, rel, Nt, S0, S1, S2);

    // out = attn . V  (heads batched, written interleaved (n, h*64+c))
    for (int w = 0; w < nw; ++w) {
      gemm(stream, attn + (size_t)w * NHEADS * Nt * Nt, Nt, (long)Nt * Nt,
           qkv + (size_t)w * Nt * 3 * EMB + 2 * EMB, 3 * EMB, 1, HD,
           t2 + (size_t)w * Nt * EMB, EMB, HD,
           Nt, HD, Nt, NHEADS, 1.f, nullptr, 0);
    }

    // proj + residual (with window unpartition for windowed blocks)
    gemm(stream, t2, EMB, 0, p.pw, EMB, 1, 0, t3, EMB, 0,
         TOKENS, EMB, EMB, 1, 1.f, p.pb, 0);
    if (!glob)
      hipLaunchKernelGGL(k_win_scatter_add, dim3(nblk(TOKENS * EMB)), dim3(TPB), 0, stream, x, t3);
    else
      hipLaunchKernelGGL(k_add_inplace, dim3(nblk(TOKENS * EMB)), dim3(TPB), 0, stream,
                         x, t3, TOKENS * EMB);

    // MLP: LN2 -> fc1(+GELU) -> fc2 -> residual
    hipLaunchKernelGGL(k_layernorm, dim3(TOKENS), dim3(TPB), 0, stream,
                       x, t1, p.n2w, p.n2b, EMB, 1e-5f);
    float* hidden = regionA;
    gemm(stream, t1, EMB, 0, p.f1w, 3072, 1, 0, hidden, 3072, 0,
         TOKENS, 3072, EMB, 1, 1.f, p.f1b, 1 /*gelu*/);
    gemm(stream, hidden, 3072, 0, p.f2w, EMB, 1, 0, t3, EMB, 0,
         TOKENS, EMB, 3072, 1, 1.f, p.f2b, 0);
    hipLaunchKernelGGL(k_add_inplace, dim3(nblk(TOKENS * EMB)), dim3(TPB), 0, stream,
                       x, t3, TOKENS * EMB);
  }

  // ---- neck ---------------------------------------------------------------
  gemm(stream, x, EMB, 0, neck_w1, 1, EMB, 0, na, 256, 0,
       TOKENS, 256, EMB, 1, 1.f, nullptr, 0);
  hipLaunchKernelGGL(k_layernorm, dim3(TOKENS), dim3(TPB), 0, stream,
                     na, nb, neck_l1w, neck_l1b, 256, 1e-6f);
  hipLaunchKernelGGL(k_im2col_neck, dim3(nblk((long)TOKENS * 6912)), dim3(TPB), 0, stream,
                     nb, regionA);
  gemm(stream, regionA, 6912, 0, neck_w2, 1, 6912, 0, na, 256, 0,
       TOKENS, 256, 6912, 1, 1.f, nullptr, 0);
  hipLaunchKernelGGL(k_ln_ch_out, dim3(TOKENS), dim3(TPB), 0, stream,
                     na, neck_l2w, neck_l2b, (float*)d_out);
}
